// ModernBertAttention_14474039788078
// MI455X (gfx1250) — matrix operand order repaired and checked by reading, not compiled, not run on hardware
//
#include <hip/hip_runtime.h>
#include <hip/hip_bf16.h>

// ---- types ---------------------------------------------------------------
typedef __bf16 bf16_t;
typedef __attribute__((ext_vector_type(16))) __bf16 v16bf;
typedef __attribute__((ext_vector_type(8)))  __bf16 v8bf;
typedef __attribute__((ext_vector_type(4)))  __bf16 v4bf;
typedef __attribute__((ext_vector_type(8)))  float  v8f;

#define B_    2
#define S_    2048
#define D_    768
#define H_    12
#define HD_   64
#define WIN_  64
#define BS_   (B_ * S_)     // 4096 rows
#define QKVN_ (3 * D_)      // 2304 cols

union FragBF { v16bf v; v8bf h[2]; };

static __device__ __forceinline__ v8f wmma_bf16(v16bf a, v16bf b, v8f c) {
  // D = A(16x32 bf16) * B(32x16 bf16) + C(16x16 f32)
  return __builtin_amdgcn_wmma_f32_16x16x32_bf16(false, a, false, b,
                                                 (short)0, c, false, false);
}

static __device__ __forceinline__ v8f vzero8() {
  v8f z = {0.f, 0.f, 0.f, 0.f, 0.f, 0.f, 0.f, 0.f};
  return z;
}

// async global->LDS copy of 16 bytes (per lane), tracked by ASYNCcnt
static __device__ __forceinline__ void async_b128(unsigned lds_off,
                                                  const void* gaddr) {
  asm volatile("global_load_async_to_lds_b128 %0, %1, off"
               :: "v"(lds_off), "v"((unsigned long long)(uintptr_t)gaddr)
               : "memory");
}
static __device__ __forceinline__ void wait_async0() {
  asm volatile("s_wait_asynccnt 0x0" ::: "memory");
}

// ---- GEMM: C[M,N] = A[M,K] * W[N,K]^T, bf16 WMMA, fp32 accumulate --------
// Block tile 128x64 (4 waves, each 32x64 -> 8 WMMA per K=32 chunk).
// A_IS_BF16: 0 -> A is fp32 (converted on load), 1 -> A is bf16.
// EPI_QKV:   0 -> store fp32 C, 1 -> scatter bf16 into Q/K/V [B,H,S,HD].
template<int A_IS_BF16, int EPI_QKV>
__global__ __launch_bounds__(128)
void gemm_wmma_kernel(const void* __restrict__ Aptr, const float* __restrict__ W,
                      float* __restrict__ Cout,
                      bf16_t* __restrict__ Qb, bf16_t* __restrict__ Kb,
                      bf16_t* __restrict__ Vb,
                      int M, int N, int K)
{
  __shared__ bf16_t sA[128][40];  // 128x32 tile, pad rows to 80B
  __shared__ bf16_t sB[64][40];

  const int tid  = threadIdx.x;
  const int lane = tid & 31;
  const int wave = tid >> 5;
  const int n0   = blockIdx.x * 64;
  const int m0   = blockIdx.y * 128;

  // WMMA fragment geometry (wave32, per CDNA5 ISA layouts)
  const int Mf  = lane & 15;             // A row / B col per lane
  const int kbA = (lane < 16) ? 0 : 8;   // A-matrix K base
  const int kbB = (lane < 16) ? 0 : 16;  // unused (B loads use kbA)
  const int rowHi = (lane >> 4) * 8;     // C-matrix M offset

  v8f acc[2][4];
#pragma unroll
  for (int mi = 0; mi < 2; ++mi)
#pragma unroll
    for (int nt = 0; nt < 4; ++nt) acc[mi][nt] = vzero8();

  for (int kk = 0; kk < K; kk += 32) {
    // stage A: 128x32 (8 float4/thread)
#pragma unroll
    for (int g = 0; g < 8; ++g) {
      int idx = g * 512 + tid * 4;
      int r = idx >> 5, c = idx & 31;
      if (A_IS_BF16) {
        const bf16_t* Ag = (const bf16_t*)Aptr;
        *(v4bf*)&sA[r][c] = *(const v4bf*)(Ag + (size_t)(m0 + r) * K + kk + c);
      } else {
        const float* Ag = (const float*)Aptr;
        const float4 a4 = *(const float4*)(Ag + (size_t)(m0 + r) * K + kk + c);
        sA[r][c + 0] = (bf16_t)a4.x; sA[r][c + 1] = (bf16_t)a4.y;
        sA[r][c + 2] = (bf16_t)a4.z; sA[r][c + 3] = (bf16_t)a4.w;
      }
    }
    // stage B: 64x32 (4 float4/thread)
#pragma unroll
    for (int g = 0; g < 4; ++g) {
      int idx = g * 512 + tid * 4;
      int r = idx >> 5, c = idx & 31;
      const float4 b4 = *(const float4*)(W + (size_t)(n0 + r) * K + kk + c);
      sB[r][c + 0] = (bf16_t)b4.x; sB[r][c + 1] = (bf16_t)b4.y;
      sB[r][c + 2] = (bf16_t)b4.z; sB[r][c + 3] = (bf16_t)b4.w;
    }
    if (kk + 32 < K) {  // gfx1250 global_prefetch_b8 for next K chunk
      const size_t esz = A_IS_BF16 ? 2 : 4;
      __builtin_prefetch((const char*)Aptr +
                         ((size_t)(m0 + tid) * K + kk + 32) * esz, 0, 1);
      __builtin_prefetch((const char*)(W + (size_t)(n0 + (tid >> 1)) * K + kk + 32),
                         0, 1);
    }
    __syncthreads();

    // A fragments: lanes 0-15 hold K {0..7,16..23}, lanes 16-31 {8..15,24..31}
    FragBF a[2];
#pragma unroll
    for (int mi = 0; mi < 2; ++mi) {
      a[mi].h[0] = *(const v8bf*)&sA[wave * 32 + mi * 16 + Mf][kbA];
      a[mi].h[1] = *(const v8bf*)&sA[wave * 32 + mi * 16 + Mf][16 + kbA];
    }
#pragma unroll
    for (int nt = 0; nt < 4; ++nt) {
      // B fragment: lane=N, K {0..7,16..23} / {8..15,24..31} like A
      FragBF b;
      b.h[0] = *(const v8bf*)&sB[nt * 16 + Mf][kbA];
      b.h[1] = *(const v8bf*)&sB[nt * 16 + Mf][16 + kbA];
#pragma unroll
      for (int mi = 0; mi < 2; ++mi)
        acc[mi][nt] = wmma_bf16(a[mi].v, b.v, acc[mi][nt]);
    }
    __syncthreads();
  }

  // epilogue: C layout VGPR r -> M = r + 8*(lane>>4), N = lane&15
#pragma unroll
  for (int mi = 0; mi < 2; ++mi) {
#pragma unroll
    for (int nt = 0; nt < 4; ++nt) {
#pragma unroll
      for (int r = 0; r < 8; ++r) {
        const int m = m0 + wave * 32 + mi * 16 + rowHi + r;
        const int n = n0 + nt * 16 + (lane & 15);
        const float val = acc[mi][nt][r];
        if (EPI_QKV) {
          const int t   = n / D_;          // 0=Q, 1=K, 2=V
          const int rem = n - t * D_;
          const int h   = rem >> 6;
          const int hd  = rem & 63;
          const int b   = m >> 11;         // S_ = 2048
          const int s   = m & (S_ - 1);
          bf16_t* dst = (t == 0) ? Qb : ((t == 1) ? Kb : Vb);
          dst[(((size_t)(b * H_ + h)) * S_ + s) * HD_ + hd] = (bf16_t)val;
        } else {
          Cout[(size_t)m * N + n] = val;
        }
      }
    }
  }
}

// ---- RoPE (in place on head-major bf16 Q,K) ------------------------------
__global__ __launch_bounds__(32)
void rope_kernel(bf16_t* __restrict__ Qb, bf16_t* __restrict__ Kb,
                 const int* __restrict__ pos_ids)
{
  const int blk = blockIdx.x;            // over B*H*S
  const int s   = blk & (S_ - 1);
  const int bh  = blk >> 11;
  const int b   = bh / H_;
  const int i   = threadIdx.x;           // pair index 0..31
  const float pos = (float)pos_ids[b * S_ + s];
  // inv_freq = theta^(-2i/HD) = exp(-(2i/HD)*ln(1e4))
  const float invf = __expf(((float)(2 * i) / (float)HD_) * -9.210340371976184f);
  float sn, c;
  __sincosf(pos * invf, &sn, &c);
  const size_t base = ((size_t)bh * S_ + s) * HD_;
  const float q1 = (float)Qb[base + i], q2 = (float)Qb[base + i + 32];
  Qb[base + i]      = (bf16_t)(q1 * c - q2 * sn);
  Qb[base + i + 32] = (bf16_t)(q2 * c + q1 * sn);
  const float k1 = (float)Kb[base + i], k2 = (float)Kb[base + i + 32];
  Kb[base + i]      = (bf16_t)(k1 * c - k2 * sn);
  Kb[base + i + 32] = (bf16_t)(k2 * c + k1 * sn);
}

// ---- sliding-window attention: one wave per (b,h,16-query tile) ----------
__global__ __launch_bounds__(32)
void attn_kernel(const bf16_t* __restrict__ Qb, const bf16_t* __restrict__ Kb,
                 const bf16_t* __restrict__ Vb, bf16_t* __restrict__ Ao)
{
  __shared__ bf16_t sQ[16][72];     // 16 x 64,  pad to 144B rows
  __shared__ bf16_t sK[160][72];    // key band x 64
  __shared__ bf16_t sVT[64][168];   // V transposed: hd x key
  __shared__ bf16_t sP[16][168];    // probabilities

  const int blk = blockIdx.x;
  const int qt  = blk & 127;               // S_/16 tiles
  const int bh  = blk >> 7;
  const int h   = bh % H_;
  const int b   = bh / H_;
  const int q0  = qt * 16;
  const int k0  = q0 - 64;                 // band covers q0-64 .. q0+79 (padded)
  const int lane = threadIdx.x;

  const size_t headBase = (size_t)bh * S_ * HD_;

  // Q tile via async global->LDS (two lanes per row, 64B each)
  {
    const int row = lane >> 1, half = (lane & 1) * 32;
    const bf16_t* src = Qb + headBase + (size_t)(q0 + row) * HD_ + half;
    const unsigned dst = (unsigned)(uintptr_t)&sQ[row][half];
#pragma unroll
    for (int j = 0; j < 4; ++j) async_b128(dst + j * 16, src + j * 8);
  }

  // K band via async global->LDS; V band loaded + transposed manually.
  // Out-of-range keys are clamped (their scores are masked to -1e9, so
  // P becomes exactly 0 and stale K/V data cannot contribute).
  for (int kk = lane; kk < 160; kk += 32) {
    int key = k0 + kk;
    int ck = key < 0 ? 0 : (key >= S_ ? S_ - 1 : key);
    const bf16_t* ksrc = Kb + headBase + (size_t)ck * HD_;
    const unsigned kdst = (unsigned)(uintptr_t)&sK[kk][0];
#pragma unroll
    for (int j = 0; j < 8; ++j) async_b128(kdst + j * 16, ksrc + j * 8);

    const bf16_t* vsrc = Vb + headBase + (size_t)ck * HD_;
#pragma unroll
    for (int j = 0; j < 16; ++j) {
      v4bf vv = *(const v4bf*)(vsrc + j * 4);
#pragma unroll
      for (int e = 0; e < 4; ++e) sVT[j * 4 + e][kk] = vv[e];
    }
  }
  wait_async0();
  __syncthreads();

  const int Mf  = lane & 15;
  const int kbA = (lane < 16) ? 0 : 8;
  const int kbB = (lane < 16) ? 0 : 16;
  const int rowHi = (lane >> 4) * 8;

  // Q fragments (HD=64 -> two K=32 chunks)
  FragBF qf[2];
#pragma unroll
  for (int c = 0; c < 2; ++c) {
    qf[c].h[0] = *(const v8bf*)&sQ[Mf][c * 32 + kbA];
    qf[c].h[1] = *(const v8bf*)&sQ[Mf][c * 32 + 16 + kbA];
  }

  // scores: 10 key tiles x (2 WMMA over HD)
  float sc[10][8];
#pragma unroll
  for (int kt = 0; kt < 10; ++kt) {
    v8f a = vzero8();
#pragma unroll
    for (int c = 0; c < 2; ++c) {
      FragBF kf;
      kf.h[0] = *(const v8bf*)&sK[kt * 16 + Mf][c * 32 + kbA];
      kf.h[1] = *(const v8bf*)&sK[kt * 16 + Mf][c * 32 + 16 + kbA];
      a = wmma_bf16(qf[c].v, kf.v, a);
    }
    const int key = k0 + kt * 16 + Mf;   // N = lane&15
#pragma unroll
    for (int r = 0; r < 8; ++r) {
      const int q = q0 + rowHi + r;
      int d = q - key; d = (d < 0) ? -d : d;
      const bool ok = (key >= 0) && (key < S_) && (d <= WIN_);
      sc[kt][r] = ok ? a[r] * 0.125f : -1e9f;  // scale = HD^-0.5
    }
  }

  // softmax per row: per-lane partials + shfl_xor across the 16-lane half
#pragma unroll
  for (int r = 0; r < 8; ++r) {
    float m = sc[0][r];
#pragma unroll
    for (int kt = 1; kt < 10; ++kt) m = fmaxf(m, sc[kt][r]);
#pragma unroll
    for (int off = 1; off < 16; off <<= 1) m = fmaxf(m, __shfl_xor(m, off, 32));
    float sum = 0.f;
#pragma unroll
    for (int kt = 0; kt < 10; ++kt) {
      const float e = __expf(sc[kt][r] - m);
      sc[kt][r] = e;
      sum += e;
    }
#pragma unroll
    for (int off = 1; off < 16; off <<= 1) sum += __shfl_xor(sum, off, 32);
    const float inv = 1.0f / sum;
    const int prow = rowHi + r;
#pragma unroll
    for (int kt = 0; kt < 10; ++kt)
      sP[prow][kt * 16 + Mf] = (bf16_t)(sc[kt][r] * inv);
  }
  __syncthreads();

  // out = P(16x160) * V(160x64): 4 N-tiles x 5 K=32 chunks
  v8f o[4];
#pragma unroll
  for (int nt = 0; nt < 4; ++nt) o[nt] = vzero8();
#pragma unroll
  for (int c = 0; c < 5; ++c) {
    FragBF p;
    p.h[0] = *(const v8bf*)&sP[Mf][c * 32 + kbA];
    p.h[1] = *(const v8bf*)&sP[Mf][c * 32 + 16 + kbA];
#pragma unroll
    for (int nt = 0; nt < 4; ++nt) {
      FragBF vf;
      vf.h[0] = *(const v8bf*)&sVT[nt * 16 + Mf][c * 32 + kbA];
      vf.h[1] = *(const v8bf*)&sVT[nt * 16 + Mf][c * 32 + 16 + kbA];
      o[nt] = wmma_bf16(p.v, vf.v, o[nt]);
    }
  }

  // store [B,S,D] bf16, col = h*64 + nt*16 + N
#pragma unroll
  for (int nt = 0; nt < 4; ++nt) {
#pragma unroll
    for (int r = 0; r < 8; ++r) {
      const int q = q0 + rowHi + r;
      Ao[((size_t)(b * S_ + q)) * D_ + h * HD_ + nt * 16 + Mf] = (bf16_t)o[nt][r];
    }
  }
}

// ---- launcher ------------------------------------------------------------
extern "C" void kernel_launch(void* const* d_in, const int* in_sizes, int n_in,
                              void* d_out, int out_size, void* d_ws, size_t ws_size,
                              hipStream_t stream) {
  (void)in_sizes; (void)n_in; (void)out_size; (void)ws_size;
  const float* X    = (const float*)d_in[0];   // [B,S,D] fp32
  const float* Wqkv = (const float*)d_in[1];   // [3D,D]  fp32
  const float* Wo   = (const float*)d_in[2];   // [D,D]   fp32
  const int*   pos  = (const int*)d_in[4];     // [B,S]   int32
  float* out = (float*)d_out;                  // [B,S,D] fp32

  const size_t nHead = (size_t)B_ * H_ * S_ * HD_;   // 3,145,728 elems
  bf16_t* Qb = (bf16_t*)d_ws;
  bf16_t* Kb = Qb + nHead;
  bf16_t* Vb = Kb + nHead;
  bf16_t* Ao = Vb + nHead;                           // [B*S, D] bf16

  // 1) fused QKV projection -> bf16 head-major Q/K/V
  gemm_wmma_kernel<0, 1><<<dim3(QKVN_ / 64, BS_ / 128), dim3(128), 0, stream>>>(
      X, Wqkv, nullptr, Qb, Kb, Vb, BS_, QKVN_, D_);
  // 2) RoPE in place on Q,K
  rope_kernel<<<dim3(B_ * H_ * S_), dim3(32), 0, stream>>>(Qb, Kb, pos);
  // 3) sliding-window attention
  attn_kernel<<<dim3(B_ * H_ * (S_ / 16)), dim3(32), 0, stream>>>(Qb, Kb, Vb, Ao);
  // 4) output projection -> fp32
  gemm_wmma_kernel<1, 0><<<dim3(D_ / 64, BS_ / 128), dim3(128), 0, stream>>>(
      Ao, Wo, out, nullptr, nullptr, nullptr, BS_, D_, D_);
}
